// ScaledDotProductAttention_4458176053409
// MI455X (gfx1250) — matrix operand order repaired and checked by reading, not compiled, not run on hardware
//
#include <hip/hip_runtime.h>

// ---------------------------------------------------------------------------
// Scaled dot product attention for MI455X (gfx1250, wave32, WMMA).
// B=2,H=16,S=2048,DK=64.  Outputs: context [B,H,S,DK] then attn_prob [B,H,S,S].
// Memory-bound on the 537MB attn_prob write -> scores live in WGP LDS (128KB),
// bf16 WMMA matmuls (independent accumulator chains), fp32 softmax with b128
// LDS ops, async LDS->global drain of attn_prob overlapped with the PV WMMAs.
// ---------------------------------------------------------------------------

typedef __attribute__((ext_vector_type(16))) __bf16 v16bf;
typedef __attribute__((ext_vector_type(8)))  float  v8f;
typedef __attribute__((ext_vector_type(4)))  float  v4f;
typedef __attribute__((ext_vector_type(2)))  float  v2f;
typedef __attribute__((ext_vector_type(4)))  int    v4i;

#if __has_builtin(__builtin_amdgcn_global_store_async_from_lds_b128) && \
    __has_builtin(__builtin_amdgcn_s_wait_asynccnt)
#define HAVE_ASYNC_LDS 1
#endif

namespace cfg {
constexpr int BH   = 32;     // B*H
constexpr int S    = 2048;
constexpr int D    = 64;
constexpr int MT   = 16;     // query rows per workgroup
constexpr int SSTR = 2052;   // padded LDS row stride (floats) -> no bank conflicts
constexpr float SCALE = 0.125f;        // 1/sqrt(64)
constexpr float NEGV  = -1.0e9f;
}

__device__ __forceinline__ __bf16 f2bf(float f) { return (__bf16)f; }  // v_cvt_pk_bf16_f32

// A-fragment K-offset for 16-bit A 16x32 (ISA 7.12.2): lane holds row m=lane&15.
__device__ __forceinline__ int a_koff(int j, int lane) {
  return (j < 4) ? ((lane < 16 ? 0 : 8) + 2 * j)
                 : ((lane < 16 ? 16 : 24) + 2 * (j - 4));
}

// A fragment from a row-major fp32 row pointer (global Q row or LDS P row).
__device__ __forceinline__ v16bf load_afrag(const float* rowp, int k0, int lane) {
  v16bf a;
#pragma unroll
  for (int j = 0; j < 8; ++j) {
    const v2f p = *(const v2f*)(rowp + k0 + a_koff(j, lane));   // 8B aligned
    a[2 * j]     = f2bf(p[0]);
    a[2 * j + 1] = f2bf(p[1]);
  }
  return a;
}

// B fragment from a V column (n = lane&15), k in table order (same as A).
__device__ __forceinline__ v16bf load_bfrag_row(const float* vc, int k0, int lane) {
  v16bf b;
#pragma unroll
  for (int j = 0; j < 8; ++j) {
    const int k = k0 + a_koff(j, lane);
    b[2 * j]     = f2bf(vc[(size_t)k * cfg::D]);
    b[2 * j + 1] = f2bf(vc[(size_t)(k + 1) * cfg::D]);
  }
  return b;
}

__global__ __launch_bounds__(256)
void sdpa_wmma_kernel(const float* __restrict__ Q, const float* __restrict__ K,
                      const float* __restrict__ V, const unsigned char* __restrict__ mask,
                      float* __restrict__ ctx_out, float* __restrict__ attn_out)
{
  using namespace cfg;
  __shared__ float sc[MT * SSTR];   // scores -> exp -> normalized probs (~128KB)
  __shared__ float part[MT * D];    // phase-3 partial accumulators (key half 1)
  __shared__ float red[MT * 16];    // row reduction scratch
  __shared__ float rowmax[MT];
  __shared__ float rowinv[MT];

  const int tid  = threadIdx.x;
  const int wave = tid >> 5;
  const int lane = tid & 31;
  const int bh   = blockIdx.x >> 7;      // 0..31
  const int qt   = blockIdx.x & 127;     // 0..127
  const int q0   = qt * MT;
  const int b    = bh >> 4;

  const float* Qb = Q + (size_t)bh * S * D;
  const float* Kb = K + (size_t)bh * S * D;
  const float* Vb = V + (size_t)bh * S * D;
  const unsigned char* Mb = mask + (size_t)b * S * S + (size_t)q0 * S;

  // ---- Phase 1: scores = mask(scale * Q K^T)  ->  LDS ---------------------
  const v16bf a0 = load_afrag(Qb + (size_t)(q0 + (lane & 15)) * D, 0,  lane);
  const v16bf a1 = load_afrag(Qb + (size_t)(q0 + (lane & 15)) * D, 32, lane);

#pragma unroll 2
  for (int t = wave; t < S / 16; t += 8) {
    const int n0 = t * 16;
    if (n0 + 128 < S)   // hint: next K tile for this wave -> global_prefetch_b8
      __builtin_prefetch(Kb + (size_t)(n0 + 128) * D, 0, 0);

    // B fragment (K^T, 32x16): lane&15 = key n0+n, elements = d in table order (a_koff).
    v16bf b0, b1;
#pragma unroll
    for (int j = 0; j < 8; ++j) {
      const float* kr = Kb + (size_t)(n0 + (lane & 15)) * D + a_koff(j, lane);
      b0[2 * j]     = f2bf(kr[0]);
      b0[2 * j + 1] = f2bf(kr[1]);
      b1[2 * j]     = f2bf(kr[32]);
      b1[2 * j + 1] = f2bf(kr[33]);
    }
    // Independent accumulator chains -> back-to-back WMMA issue, no RAW gap.
    v8f c0 = {}, c1 = {};
    c0 = __builtin_amdgcn_wmma_f32_16x16x32_bf16(false, a0, false, b0, (short)0, c0, false, false);
    c1 = __builtin_amdgcn_wmma_f32_16x16x32_bf16(false, a1, false, b1, (short)0, c1, false, false);
    const v8f c = c0 + c1;

    // C layout: vgpr i, lanes 0-15 -> (m=i, n=lane); lanes 16-31 -> (m=8+i, n=lane-16)
    const int n  = lane & 15;
    const int mb = (lane >> 4) << 3;
#pragma unroll
    for (int i = 0; i < 8; ++i) {
      const int m = mb + i;
      float v = c[i] * SCALE;
      if (Mb[(size_t)m * S + n0 + n]) v = NEGV;
      sc[m * SSTR + n0 + n] = v;
    }
  }
  __syncthreads();

  // ---- Phase 2: fp32 row softmax in LDS (b128 vectorized) -----------------
  const int row = tid >> 4, l16 = tid & 15;   // 16 threads per query row
  const int cbase = row * SSTR + l16 * 4;     // lanes stride 16B -> 64 banks covered
  float pm = -3.4e38f;
#pragma unroll 4
  for (int j = 0; j < S / 64; ++j) {
    const v4f v = *(const v4f*)&sc[cbase + 64 * j];             // ds_load_b128
    pm = fmaxf(pm, fmaxf(fmaxf(v[0], v[1]), fmaxf(v[2], v[3])));
  }
  red[row * 16 + l16] = pm;
  __syncthreads();
  if (l16 == 0) {
    float m = red[row * 16];
#pragma unroll
    for (int j = 1; j < 16; ++j) m = fmaxf(m, red[row * 16 + j]);
    rowmax[row] = m;
  }
  __syncthreads();

  {
    const float rm = rowmax[row];
    float ps = 0.f;
#pragma unroll 4
    for (int j = 0; j < S / 64; ++j) {
      v4f* p = (v4f*)&sc[cbase + 64 * j];
      v4f v = *p;
      v[0] = __expf(v[0] - rm);
      v[1] = __expf(v[1] - rm);
      v[2] = __expf(v[2] - rm);
      v[3] = __expf(v[3] - rm);
      *p = v;                                                   // ds_store_b128
      ps += v[0] + v[1] + v[2] + v[3];
    }
    red[row * 16 + l16] = ps;
  }
  __syncthreads();
  if (l16 == 0) {
    float s = 0.f;
#pragma unroll
    for (int j = 0; j < 16; ++j) s += red[row * 16 + j];
    rowinv[row] = 1.0f / s;
  }
  __syncthreads();

  // Normalize probabilities in place (phase 3 consumes normalized P).
#pragma unroll 4
  for (int j = 0; j < (MT * S / 4) / 256; ++j) {
    const int c  = tid + 256 * j;
    const int r  = c >> 9;                                      // S/4 = 512 chunks/row
    const int c4 = (c & 511) * 4;
    v4f* p = (v4f*)&sc[r * SSTR + c4];
    *p = *p * rowinv[r];
  }
  __syncthreads();

  // ---- attn_prob writeback: 537MB stream, async LDS->global ---------------
  {
    float* ap = attn_out + ((size_t)bh * S + q0) * S;
#ifdef HAVE_ASYNC_LDS
    // 16B per lane per issue; ASYNCcnt-tracked, overlaps with the PV WMMAs.
#pragma unroll
    for (int j = 0; j < (MT * S / 4) / 256; ++j) {     // 32 issues per thread
      const int c  = tid + 256 * j;
      const int r  = c >> 9;
      const int c4 = (c & 511) * 4;
      __builtin_amdgcn_global_store_async_from_lds_b128(
          (v4i*)(ap + (size_t)r * S + c4),
          (v4i*)&sc[r * SSTR + c4], 0, 0);
    }
#else
    for (int j = 0; j < (MT * S / 4) / 256; ++j) {     // fallback: b128 stores
      const int c  = tid + 256 * j;
      const int r  = c >> 9;
      const int c4 = (c & 511) * 4;
      *(v4f*)(ap + (size_t)r * S + c4) = *(const v4f*)&sc[r * SSTR + c4];
    }
#endif
  }

  // ---- Phase 3: context = P x V  (bf16 WMMA, dual accumulator chains) -----
  const int ct = wave & 3;     // 16-wide d tile
  const int kh = wave >> 2;    // key half: 0 -> [0,1024), 1 -> [1024,2048)
  const float* scrow = &sc[(lane & 15) * SSTR];
  v8f acc0 = {}, acc1 = {};
  for (int kb = 0; kb < 32; kb += 2) {
    const int k0a = kh * 1024 + kb * 32;
    const int k0b = k0a + 32;
    const v16bf pa0 = load_afrag(scrow, k0a, lane);             // ds_load_b64 pairs
    const v16bf vb0 = load_bfrag_row(Vb + ct * 16 + (lane & 15), k0a, lane);
    const v16bf pa1 = load_afrag(scrow, k0b, lane);
    const v16bf vb1 = load_bfrag_row(Vb + ct * 16 + (lane & 15), k0b, lane);
    acc0 = __builtin_amdgcn_wmma_f32_16x16x32_bf16(false, pa0, false, vb0, (short)0, acc0, false, false);
    acc1 = __builtin_amdgcn_wmma_f32_16x16x32_bf16(false, pa1, false, vb1, (short)0, acc1, false, false);
  }
  const v8f acc = acc0 + acc1;

  const int n   = lane & 15;
  const int mb2 = (lane >> 4) << 3;
  if (kh == 1) {
#pragma unroll
    for (int i = 0; i < 8; ++i)
      part[(mb2 + i) * D + ct * 16 + n] = acc[i];
  }
  __syncthreads();
  if (kh == 0) {
    float* co = ctx_out + ((size_t)bh * S + q0) * D;
#pragma unroll
    for (int i = 0; i < 8; ++i) {
      const int m = mb2 + i;
      co[(size_t)m * D + ct * 16 + n] = acc[i] + part[m * D + ct * 16 + n];
    }
  }

#ifdef HAVE_ASYNC_LDS
  __builtin_amdgcn_s_wait_asynccnt(0);   // drain attn_prob stores before exit
#endif
}

extern "C" void kernel_launch(void* const* d_in, const int* in_sizes, int n_in,
                              void* d_out, int out_size, void* d_ws, size_t ws_size,
                              hipStream_t stream) {
  using namespace cfg;
  const float* Q = (const float*)d_in[0];
  const float* K = (const float*)d_in[1];
  const float* V = (const float*)d_in[2];
  const unsigned char* mask = (const unsigned char*)d_in[3];  // jax bool -> 1 byte

  float* out = (float*)d_out;
  float* ctx = out;                                  // [B,H,S,DK]
  float* ap  = out + (size_t)BH * S * D;             // [B,H,S,S]

  dim3 grid(BH * (S / MT));   // 32 * 128 = 4096 workgroups
  dim3 block(256);            // 8 wave32 per WGP
  sdpa_wmma_kernel<<<grid, block, 0, stream>>>(Q, K, V, mask, ctx, ap);
}
